// IETNet_42099269435747
// MI455X (gfx1250) — matrix operand order repaired and checked by reading, not compiled, not run on hardware
//
#include <hip/hip_runtime.h>
#include <math.h>

// ---------------------------------------------------------------------------
// Problem constants (from reference)
// ---------------------------------------------------------------------------
#define B_      16
#define T_      64
#define NPTS    512
#define C_IN    4
#define NPOINT  128
#define NSAMPLE 8
#define R2      16.0f            // RADIUS^2
#define D_      512
#define NH      8
#define DK      64
#define NL      8
#define M_ROWS  (B_ * T_)        // 1024 token rows
#define BN_INV  0.9999950000374997f      // 1/sqrt(1+1e-5)
#define LN_EPS  1e-5f
#define SQRT_D  22.62741699796952f       // sqrt(512)
#define PE_C    0.013491709529261988f    // ln(1000)/512

typedef _Float16 v16h __attribute__((ext_vector_type(16)));
typedef _Float16 v8h  __attribute__((ext_vector_type(8)));
typedef float    v8f  __attribute__((ext_vector_type(8)));

// ---------------------------------------------------------------------------
// WMMA fragment loaders (layouts per CDNA5 ISA 7.12.2, wave32)
// A (16x32 f16) row-major in LDS, ldk halfs per row:
//   lanes 0-15 : M=lane,    K = {0..7} in v0..3,  {16..23} in v4..7
//   lanes 16-31: M=lane-16, K = {8..15} in v0..3, {24..31} in v4..7
// => two 16-byte contiguous chunks -> 2x ds_load_b128.
// B (32x16 f16) staged transposed as [n][k] in LDS:
//   lanes 0-15 : N=lane,    K = {0..7},{16..23} ; lanes 16-31: N=lane-16, K = {8..15},{24..31}
// => two 16-byte contiguous chunks -> 2x ds_load_b128.
// ---------------------------------------------------------------------------
static __device__ __forceinline__ v16h frag_a(const _Float16* base, int ldk,
                                              int row0, int k0, int lane) {
  int r = lane & 15, hi = lane >> 4;
  const _Float16* p = base + (row0 + r) * ldk + k0 + hi * 8;
  v8h lo = *(const v8h*)(p);
  v8h hh = *(const v8h*)(p + 16);
  return __builtin_shufflevector(lo, hh, 0,1,2,3,4,5,6,7,8,9,10,11,12,13,14,15);
}

static __device__ __forceinline__ v16h frag_b(const _Float16* base, int ldk,
                                              int n0, int k0, int lane) {
  int r = lane & 15, hi = lane >> 4;
  const _Float16* p = base + (n0 + r) * ldk + k0 + hi * 8;
  v8h lo = *(const v8h*)(p);
  v8h hh = *(const v8h*)(p + 16);
  return __builtin_shufflevector(lo, hh, 0,1,2,3,4,5,6,7,8,9,10,11,12,13,14,15);
}

static __device__ __forceinline__ v8f wmma_f16(v16h a, v16h b, v8f c) {
  return __builtin_amdgcn_wmma_f32_16x16x32_f16(false, a, false, b,
                                                (short)0, c, false, false);
}

// ===========================================================================
// Kernel 1: per-cloud PointNet (FPS + ball query + conv1 + WMMA conv2 + max)
// grid = M (1024 clouds), block = 256 (8 waves)
// ===========================================================================
__global__ __launch_bounds__(256)
void pointnet_kernel(const float* __restrict__ xyz,
                     const float* __restrict__ c1_w, const float* __restrict__ c1_b,
                     const float* __restrict__ bn1_g, const float* __restrict__ bn1_b,
                     const float* __restrict__ c2_w, const float* __restrict__ c2_b,
                     const float* __restrict__ bn2_g, const float* __restrict__ bn2_b,
                     float* __restrict__ per_cloud) {
  __shared__ float    s_pts[NPTS * 3];       // 6 KB
  __shared__ float    s_feat[NPTS];          // 2 KB
  __shared__ float    s_dist[NPTS];          // 2 KB
  __shared__ float    s_rv[256];
  __shared__ int      s_ri[256];
  __shared__ float    s_cen[NPOINT * 3];     // 1.5 KB
  __shared__ int      s_gidx[NPOINT * NSAMPLE]; // 4 KB
  __shared__ _Float16 s_h1[128 * 64];        // 16 KB  conv1 activations (chunk)
  __shared__ _Float16 s_w2[128 * 64];        // 16 KB  c2_w is [p][o] == [n][k]
  __shared__ float    s_w1[64 * 4];
  __shared__ float    s_b1[64], s_s1[64], s_t1[64];
  __shared__ float    s_b2[128], s_s2[128], s_t2[128];
  __shared__ unsigned s_max[128];

  const int m    = blockIdx.x;
  const int tid  = threadIdx.x;
  const int wv   = tid >> 5;
  const int lane = tid & 31;

  // ---- load points, features, weight caches -------------------------------
  for (int p = tid; p < NPTS; p += 256) {
    const float* src = xyz + ((size_t)m * NPTS + p) * C_IN;
    s_pts[p * 3 + 0] = src[0];
    s_pts[p * 3 + 1] = src[1];
    s_pts[p * 3 + 2] = src[2];
    s_feat[p]        = src[3];
    s_dist[p]        = 1e10f;
  }
  s_w1[tid] = c1_w[tid];                       // 256 = 64*4
  if (tid < 64) {
    s_b1[tid] = c1_b[tid];
    s_s1[tid] = bn1_g[tid] * BN_INV;
    s_t1[tid] = bn1_b[tid];
  }
  if (tid < 128) {
    s_b2[tid]  = c2_b[tid];
    s_s2[tid]  = bn2_g[tid] * BN_INV;
    s_t2[tid]  = bn2_b[tid];
    s_max[tid] = 0u;
  }
  for (int i = tid; i < 128 * 64; i += 256)
    s_w2[i] = (_Float16)c2_w[i];
  __syncthreads();

  // ---- farthest point sampling -------------------------------------------
  int far = 0;
  for (int s = 0; s < NPOINT; ++s) {
    float cx = s_pts[far * 3 + 0];
    float cy = s_pts[far * 3 + 1];
    float cz = s_pts[far * 3 + 2];
    if (tid == 0) { s_cen[s * 3 + 0] = cx; s_cen[s * 3 + 1] = cy; s_cen[s * 3 + 2] = cz; }
    float bestv = -1.0f; int besti = 0;
    for (int p = tid; p < NPTS; p += 256) {
      float dx = s_pts[p * 3 + 0] - cx;
      float dy = s_pts[p * 3 + 1] - cy;
      float dz = s_pts[p * 3 + 2] - cz;
      float d  = dx * dx + dy * dy + dz * dz;
      float nd = fminf(s_dist[p], d);
      s_dist[p] = nd;
      if (nd > bestv) { bestv = nd; besti = p; }
    }
    s_rv[tid] = bestv; s_ri[tid] = besti;
    __syncthreads();
    for (int st = 128; st > 0; st >>= 1) {
      if (tid < st && s_rv[tid + st] > s_rv[tid]) {
        s_rv[tid] = s_rv[tid + st]; s_ri[tid] = s_ri[tid + st];
      }
      __syncthreads();
    }
    far = s_ri[0];
    __syncthreads();
  }

  // ---- ball query (first NSAMPLE in-radius indices, pad with first) ------
  if (tid < NPOINT) {
    const int s = tid;
    float cx = s_cen[s * 3 + 0], cy = s_cen[s * 3 + 1], cz = s_cen[s * 3 + 2];
    int cnt = 0, first = -1;
    for (int p = 0; p < NPTS; ++p) {
      float dx = s_pts[p * 3 + 0] - cx;
      float dy = s_pts[p * 3 + 1] - cy;
      float dz = s_pts[p * 3 + 2] - cz;
      float d  = dx * dx + dy * dy + dz * dz;
      if (d <= R2) {
        if (first < 0) first = p;
        if (cnt < NSAMPLE) s_gidx[s * NSAMPLE + cnt] = p;
        ++cnt;
      }
    }
    if (first < 0) first = 0;
    for (int j = cnt; j < NSAMPLE; ++j) s_gidx[s * NSAMPLE + j] = first;
  }
  __syncthreads();

  // ---- conv1 (VALU, tiny K=4) then conv2 via WMMA, max-pool --------------
  const int r  = lane & 15;
  const int hi = lane >> 4;
  float rmax[8];
#pragma unroll
  for (int nt = 0; nt < 8; ++nt) rmax[nt] = 0.0f;

  for (int cc = 0; cc < 8; ++cc) {            // 8 chunks of 128 samples
    {                                          // conv1 -> s_h1 (f16)
      int j    = tid >> 1;                     // sample-in-chunk 0..127
      int o0   = (tid & 1) * 32;
      int samp = cc * 128 + j;
      int s    = samp >> 3, kk = samp & 7;
      int idx  = s_gidx[s * NSAMPLE + kk];
      float gx = s_pts[idx * 3 + 0] - s_cen[s * 3 + 0];
      float gy = s_pts[idx * 3 + 1] - s_cen[s * 3 + 1];
      float gz = s_pts[idx * 3 + 2] - s_cen[s * 3 + 2];
      float gf = s_feat[idx];
#pragma unroll 8
      for (int o = o0; o < o0 + 32; ++o) {
        float acc = gx * s_w1[o * 4 + 0] + gy * s_w1[o * 4 + 1] +
                    gz * s_w1[o * 4 + 2] + gf * s_w1[o * 4 + 3] + s_b1[o];
        float val = fmaxf(acc * s_s1[o] + s_t1[o], 0.0f);
        s_h1[j * 64 + o] = (_Float16)val;
      }
    }
    __syncthreads();

    // WMMA: wave's 16-row strip x all 128 output channels, K = 64
    v16h a0 = frag_a(s_h1, 64, wv * 16, 0,  lane);
    v16h a1 = frag_a(s_h1, 64, wv * 16, 32, lane);
#pragma unroll
    for (int nt = 0; nt < 8; ++nt) {
      v16h b0 = frag_b(s_w2, 64, nt * 16, 0,  lane);
      v16h b1 = frag_b(s_w2, 64, nt * 16, 32, lane);
      v8f acc = {};
      acc = wmma_f16(a0, b0, acc);
      acc = wmma_f16(a1, b1, acc);
      int   n  = nt * 16 + r;
      float bb = s_b2[n], sc = s_s2[n], sh = s_t2[n];
#pragma unroll
      for (int rr = 0; rr < 8; ++rr) {
        float val = fmaxf((acc[rr] + bb) * sc + sh, 0.0f);
        rmax[nt]  = fmaxf(rmax[nt], val);
      }
    }
    __syncthreads();
  }

#pragma unroll
  for (int nt = 0; nt < 8; ++nt)
    atomicMax(&s_max[nt * 16 + r], __float_as_uint(rmax[nt]));  // ds_max_u32
  __syncthreads();
  if (tid < 128) per_cloud[m * 128 + tid] = __uint_as_float(s_max[tid]);
}

// ===========================================================================
// Kernel 2: generic WMMA GEMM  C = A(MxK,f32) * W(KxN,f32) + bias
// tile 64x64x32, block = 128 (4 waves), grid = (M/64, N/64)
// ===========================================================================
__global__ __launch_bounds__(128)
void gemm_bias_kernel(const float* __restrict__ A, const float* __restrict__ W,
                      const float* __restrict__ bias, float* __restrict__ C,
                      int M, int N, int K) {
  __shared__ _Float16 sA[64 * 32];   // [row][k]
  __shared__ _Float16 sB[64 * 32];   // [n][k]  (W staged transposed)

  const int tid  = threadIdx.x;
  const int wv   = tid >> 5;
  const int lane = tid & 31;
  const int m0   = blockIdx.x * 64;
  const int n0   = blockIdx.y * 64;
  const int r    = lane & 15;
  const int hi   = lane >> 4;

  v8f acc[4] = {};

  for (int k0 = 0; k0 < K; k0 += 32) {
    {  // stage A: 64 rows x 32 k
      int rr = tid >> 1;
      int kk = (tid & 1) * 16;
      const float* src = A + (size_t)(m0 + rr) * K + k0 + kk;
      _Float16* dst = sA + rr * 32 + kk;
      if (k0 + 32 < K) __builtin_prefetch(src + 32, 0, 3);
#pragma unroll
      for (int i = 0; i < 16; i += 4) {
        float4 f = *(const float4*)(src + i);
        dst[i + 0] = (_Float16)f.x; dst[i + 1] = (_Float16)f.y;
        dst[i + 2] = (_Float16)f.z; dst[i + 3] = (_Float16)f.w;
      }
    }
    {  // stage W transposed: W[k][n] -> sB[n][k]
      int kk = tid >> 2;
      int nn = (tid & 3) * 16;
      const float* src = W + (size_t)(k0 + kk) * N + n0 + nn;
#pragma unroll
      for (int i = 0; i < 16; i += 4) {
        float4 f = *(const float4*)(src + i);
        sB[(nn + i + 0) * 32 + kk] = (_Float16)f.x;
        sB[(nn + i + 1) * 32 + kk] = (_Float16)f.y;
        sB[(nn + i + 2) * 32 + kk] = (_Float16)f.z;
        sB[(nn + i + 3) * 32 + kk] = (_Float16)f.w;
      }
    }
    __syncthreads();

    v16h af = frag_a(sA, 32, wv * 16, 0, lane);
#pragma unroll
    for (int nt = 0; nt < 4; ++nt) {
      v16h bf = frag_b(sB, 32, nt * 16, 0, lane);
      acc[nt] = wmma_f16(af, bf, acc[nt]);
    }
    __syncthreads();
  }

#pragma unroll
  for (int nt = 0; nt < 4; ++nt) {
    int col = n0 + nt * 16 + r;
    float bc = bias ? bias[col] : 0.0f;
#pragma unroll
    for (int rr = 0; rr < 8; ++rr) {
      int row = m0 + wv * 16 + rr + hi * 8;
      C[(size_t)row * N + col] = acc[nt][rr] + bc;
    }
  }
}

// ===========================================================================
// Kernel 3: multi-head attention for one (b,h): 64x64 tiles, all WMMA
// grid = (B, NH), block = 128 (4 waves)
// ===========================================================================
__global__ __launch_bounds__(128)
void attn_kernel(const float* __restrict__ q, const float* __restrict__ k,
                 const float* __restrict__ v, float* __restrict__ o) {
  __shared__ _Float16 sQ[64 * 64];   // [t][d]
  __shared__ _Float16 sK[64 * 64];   // [t][d]  == B layout [n][k] for Q*K^T
  __shared__ _Float16 sVT[64 * 64];  // [d][t]  == B layout [n][k] for P*V
  __shared__ _Float16 sP[64 * 64];   // softmax probs (f16)
  __shared__ float    sS[64 * 64];   // scores (f32)

  const int b    = blockIdx.x;
  const int h    = blockIdx.y;
  const int tid  = threadIdx.x;
  const int wv   = tid >> 5;
  const int lane = tid & 31;
  const int r    = lane & 15;
  const int hi   = lane >> 4;

  {  // load Q, K, V^T (f32 -> f16)
    int row = tid >> 1;
    int d0  = (tid & 1) * 32;
    size_t base = ((size_t)(b * T_ + row)) * D_ + h * DK + d0;
#pragma unroll
    for (int i = 0; i < 32; i += 4) {
      float4 fq = *(const float4*)(q + base + i);
      float4 fk = *(const float4*)(k + base + i);
      float4 fv = *(const float4*)(v + base + i);
      sQ[row * 64 + d0 + i + 0] = (_Float16)fq.x;
      sQ[row * 64 + d0 + i + 1] = (_Float16)fq.y;
      sQ[row * 64 + d0 + i + 2] = (_Float16)fq.z;
      sQ[row * 64 + d0 + i + 3] = (_Float16)fq.w;
      sK[row * 64 + d0 + i + 0] = (_Float16)fk.x;
      sK[row * 64 + d0 + i + 1] = (_Float16)fk.y;
      sK[row * 64 + d0 + i + 2] = (_Float16)fk.z;
      sK[row * 64 + d0 + i + 3] = (_Float16)fk.w;
      sVT[(d0 + i + 0) * 64 + row] = (_Float16)fv.x;
      sVT[(d0 + i + 1) * 64 + row] = (_Float16)fv.y;
      sVT[(d0 + i + 2) * 64 + row] = (_Float16)fv.z;
      sVT[(d0 + i + 3) * 64 + row] = (_Float16)fv.w;
    }
  }
  __syncthreads();

  // S = Q * K^T / sqrt(dk)
#pragma unroll
  for (int nt = 0; nt < 4; ++nt) {
    v8f acc = {};
#pragma unroll
    for (int ks = 0; ks < 2; ++ks) {
      v16h af = frag_a(sQ, 64, wv * 16, ks * 32, lane);
      v16h bf = frag_b(sK, 64, nt * 16, ks * 32, lane);
      acc = wmma_f16(af, bf, acc);
    }
#pragma unroll
    for (int rr = 0; rr < 8; ++rr)
      sS[(wv * 16 + rr + hi * 8) * 64 + nt * 16 + r] = acc[rr] * 0.125f;
  }
  __syncthreads();

  // softmax rows (threads 0..63)
  if (tid < 64) {
    float mx = -1e30f;
#pragma unroll 8
    for (int j = 0; j < 64; ++j) mx = fmaxf(mx, sS[tid * 64 + j]);
    float sum = 0.0f;
#pragma unroll 8
    for (int j = 0; j < 64; ++j) {
      float e = __expf(sS[tid * 64 + j] - mx);
      sS[tid * 64 + j] = e;
      sum += e;
    }
    float inv = 1.0f / sum;
#pragma unroll 8
    for (int j = 0; j < 64; ++j)
      sP[tid * 64 + j] = (_Float16)(sS[tid * 64 + j] * inv);
  }
  __syncthreads();

  // O = P * V
#pragma unroll
  for (int nt = 0; nt < 4; ++nt) {
    v8f acc = {};
#pragma unroll
    for (int ks = 0; ks < 2; ++ks) {
      v16h af = frag_a(sP, 64, wv * 16, ks * 32, lane);
      v16h bf = frag_b(sVT, 64, nt * 16, ks * 32, lane);
      acc = wmma_f16(af, bf, acc);
    }
#pragma unroll
    for (int rr = 0; rr < 8; ++rr) {
      int row = wv * 16 + rr + hi * 8;
      o[((size_t)(b * T_ + row)) * D_ + h * DK + nt * 16 + r] = acc[rr];
    }
  }
}

// ===========================================================================
// Kernel 4: layernorm (wave per row of 512). modes:
//   0: out = LN(x [+ res]) * g + b
//   1: relu(LN(x) * g + b)
//   3: LN(x)*g+b -> *sqrt(D) + positional encoding (token stage)
// grid = M/4, block = 128
// ===========================================================================
__global__ __launch_bounds__(128)
void ln_kernel(const float* __restrict__ x, const float* __restrict__ res,
               const float* __restrict__ g, const float* __restrict__ bta,
               float* __restrict__ out, int mode) {
  const int row  = blockIdx.x * 4 + (threadIdx.x >> 5);
  const int lane = threadIdx.x & 31;
  const float* xp = x + (size_t)row * D_;

  float vals[16];
  float s = 0.0f, ss = 0.0f;
#pragma unroll
  for (int i = 0; i < 16; ++i) {
    int c = lane + i * 32;
    float t = xp[c];
    if (res) t += res[(size_t)row * D_ + c];
    vals[i] = t;
    s += t; ss += t * t;
  }
#pragma unroll
  for (int off = 16; off > 0; off >>= 1) {
    s  += __shfl_xor(s,  off, 32);
    ss += __shfl_xor(ss, off, 32);
  }
  float mean = s * (1.0f / D_);
  float var  = ss * (1.0f / D_) - mean * mean;
  float inv  = rsqrtf(var + LN_EPS);

  const int t_idx = row & (T_ - 1);
#pragma unroll
  for (int i = 0; i < 16; ++i) {
    int c = lane + i * 32;
    float y = (vals[i] - mean) * inv * g[c] + bta[c];
    if (mode == 1) y = fmaxf(y, 0.0f);
    if (mode == 3) {
      float div = __expf(-(float)(c & ~1) * PE_C);
      float pe  = (c & 1) ? __cosf((float)t_idx * div) : __sinf((float)t_idx * div);
      y = y * SQRT_D + pe;
    }
    out[(size_t)row * D_ + c] = y;
  }
}

// ===========================================================================
// Host orchestration
// ===========================================================================
static inline void gemm(const float* A, const float* W, const float* bias,
                        float* C, int M, int N, int K, hipStream_t s) {
  dim3 grid(M / 64, N / 64);
  gemm_bias_kernel<<<grid, 128, 0, s>>>(A, W, bias, C, M, N, K);
}
static inline void ln(const float* x, const float* res, const float* g,
                      const float* b, float* out, int mode, hipStream_t s) {
  ln_kernel<<<M_ROWS / 4, 128, 0, s>>>(x, res, g, b, out, mode);
}

extern "C" void kernel_launch(void* const* d_in, const int* in_sizes, int n_in,
                              void* d_out, int out_size, void* d_ws, size_t ws_size,
                              hipStream_t stream) {
  (void)in_sizes; (void)n_in; (void)out_size; (void)ws_size;
  const float* xyz    = (const float*)d_in[0];
  const float* c1_w   = (const float*)d_in[1];
  const float* c1_b   = (const float*)d_in[2];
  const float* bn1_g  = (const float*)d_in[3];
  const float* bn1_b  = (const float*)d_in[4];
  const float* c2_w   = (const float*)d_in[5];
  const float* c2_b   = (const float*)d_in[6];
  const float* bn2_g  = (const float*)d_in[7];
  const float* bn2_b  = (const float*)d_in[8];
  const float* fc_w   = (const float*)d_in[9];
  const float* fc_b   = (const float*)d_in[10];
  const float* tn_g   = (const float*)d_in[11];
  const float* tn_b   = (const float*)d_in[12];
  const float* attn_w = (const float*)d_in[13];
  const float* attn_b = (const float*)d_in[14];
  const float* ff_w   = (const float*)d_in[15];
  const float* ff_b   = (const float*)d_in[16];
  const float* ln_g   = (const float*)d_in[17];
  const float* ln_b   = (const float*)d_in[18];
  const float* d1_w   = (const float*)d_in[19];
  const float* d1_b   = (const float*)d_in[20];
  const float* d2_w   = (const float*)d_in[21];
  const float* d2_b   = (const float*)d_in[22];
  const float* d3_w   = (const float*)d_in[23];
  const float* d3_b   = (const float*)d_in[24];
  const float* dln_g  = (const float*)d_in[25];
  const float* dln_b  = (const float*)d_in[26];

  // workspace layout (floats): ~17.3 MB total
  float* ws = (float*)d_ws;
  const size_t ROWSZ = (size_t)M_ROWS * D_;   // 524288
  float* per_cloud = ws;                       // 1024*128
  float* xb = per_cloud + (size_t)M_ROWS * 128;
  float* qb = xb + ROWSZ;
  float* kb = qb + ROWSZ;
  float* vb = kb + ROWSZ;
  float* ob = vb + ROWSZ;
  float* ab = ob + ROWSZ;
  float* t1 = ab + ROWSZ;
  float* t2 = t1 + ROWSZ;

  // ---- per-cloud geometry + PointNet -------------------------------------
  pointnet_kernel<<<M_ROWS, 256, 0, stream>>>(xyz, c1_w, c1_b, bn1_g, bn1_b,
                                              c2_w, c2_b, bn2_g, bn2_b, per_cloud);

  // ---- token projection + LN + *sqrt(D) + PE -----------------------------
  gemm(per_cloud, fc_w, fc_b, t1, M_ROWS, D_, 128, stream);
  ln(t1, nullptr, tn_g, tn_b, xb, 3, stream);

  // ---- transformer encoder ------------------------------------------------
  const size_t DD = (size_t)D_ * D_;
  for (int l = 0; l < NL; ++l) {
    const float* aw = attn_w + (size_t)l * 4 * DD;
    const float* abv = attn_b + (size_t)l * 4 * D_;
    const float* fw = ff_w + (size_t)l * 3 * DD;
    const float* fb = ff_b + (size_t)l * 3 * D_;
    const float* lg = ln_g + (size_t)l * 4 * D_;
    const float* lb = ln_b + (size_t)l * 4 * D_;

    gemm(xb, aw + 0 * DD, abv + 0 * D_, qb, M_ROWS, D_, D_, stream);
    gemm(xb, aw + 1 * DD, abv + 1 * D_, kb, M_ROWS, D_, D_, stream);
    gemm(xb, aw + 2 * DD, abv + 2 * D_, vb, M_ROWS, D_, D_, stream);
    attn_kernel<<<dim3(B_, NH), 128, 0, stream>>>(qb, kb, vb, ob);
    gemm(ob, aw + 3 * DD, abv + 3 * D_, ab, M_ROWS, D_, D_, stream);

    gemm(ab, fw + 0 * DD, fb + 0 * D_, t1, M_ROWS, D_, D_, stream);
    ln(t1, nullptr, lg + 0 * D_, lb + 0 * D_, t2, 1, stream);
    gemm(t2, fw + 1 * DD, fb + 1 * D_, t1, M_ROWS, D_, D_, stream);
    ln(t1, nullptr, lg + 1 * D_, lb + 1 * D_, t2, 1, stream);
    gemm(t2, fw + 2 * DD, fb + 2 * D_, t1, M_ROWS, D_, D_, stream);
    ln(t1, ab, lg + 2 * D_, lb + 2 * D_, t2, 0, stream);        // LN(h + a)
    ln(t2, nullptr, lg + 3 * D_, lb + 3 * D_, xb, 0, stream);   // LN(ff)
  }

  // ---- decoder ------------------------------------------------------------
  gemm(xb, d1_w, d1_b, t1, M_ROWS, D_, D_, stream);
  ln(t1, nullptr, dln_g + 0 * D_, dln_b + 0 * D_, t2, 1, stream);
  gemm(t2, d2_w, d2_b, t1, M_ROWS, D_, D_, stream);
  ln(t1, nullptr, dln_g + 1 * D_, dln_b + 1 * D_, t2, 1, stream);
  gemm(t2, d3_w, d3_b, (float*)d_out, M_ROWS, 128, D_, stream);
}
